// GAT_GCN_72146860638341
// MI455X (gfx1250) — matrix operand order repaired and checked by reading, not compiled, not run on hardware
//
#include <hip/hip_runtime.h>
#include <hip/hip_bf16.h>

// ---------------------------------------------------------------------------
// Types for CDNA5 WMMA
// ---------------------------------------------------------------------------
typedef __attribute__((ext_vector_type(16))) __bf16 v16bf;
typedef __attribute__((ext_vector_type(8)))  float  v8f;
typedef __attribute__((ext_vector_type(4)))  unsigned int uiv4;

#define NN   16384     // nodes
#define NE   65536     // edges (without self loops)
#define E2   (NE + NN) // edges incl. self loops
#define NB   512       // graphs / batch
#define NPG  (NN / NB) // nodes per graph (contiguous by construction)

// ---------------------------------------------------------------------------
// helpers
// ---------------------------------------------------------------------------
__device__ __forceinline__ int imin(int a, int b) { return a < b ? a : b; }
__device__ __forceinline__ unsigned short f2bf(float f) {
  unsigned u = __float_as_uint(f);
  u += 0x7FFFu + ((u >> 16) & 1u);           // round-to-nearest-even
  return (unsigned short)(u >> 16);
}
__device__ __forceinline__ unsigned fflip(float f) {   // monotone float->uint
  unsigned u = __float_as_uint(f);
  return u ^ ((u >> 31) ? 0xFFFFFFFFu : 0x80000000u);
}
__device__ __forceinline__ float funflip(unsigned u) {
  return __uint_as_float(u ^ ((u >> 31) ? 0x80000000u : 0xFFFFFFFFu));
}

__global__ void fill_f32(float* p, float v, long long n) {
  long long i = (long long)blockIdx.x * blockDim.x + threadIdx.x;
  if (i < n) p[i] = v;
}
__global__ void fill_u32(unsigned* p, unsigned v, long long n) {
  long long i = (long long)blockIdx.x * blockDim.x + threadIdx.x;
  if (i < n) p[i] = v;
}

// Weight pre-pass: W[K,N] f32 -> Bt[Npad,Kp] bf16, transposed + zero padded so
// the GEMM B-path never needs bounds checks and stages pure bf16 16B blocks.
__global__ void wt_bf16t(const float* __restrict__ W, unsigned short* __restrict__ Bt,
                         int K, int N, int Kp, long long total) {
  long long idx = (long long)blockIdx.x * blockDim.x + threadIdx.x;
  if (idx >= total) return;
  int n = (int)(idx / Kp), k = (int)(idx % Kp);
  float v = (n < N && k < K) ? W[(size_t)k * N + n] : 0.f;
  Bt[idx] = f2bf(v);
}

// ---------------------------------------------------------------------------
// GEMM staging helpers
// ---------------------------------------------------------------------------
// A tile: 128 rows x 32 k, f32 -> bf16, lane = contiguous k (coalesced)
__device__ __forceinline__
void stageA(const float* __restrict__ A, unsigned short* dst,
            int r0, int kb, int M, int K, int arow, int acol, bool fullM) {
  if (fullM && (kb + 32 <= K)) {
    const float* ap = A + (size_t)(r0 + arow) * K + kb + acol;
#pragma unroll
    for (int it = 0; it < 16; ++it)
      dst[(arow + it * 8) * 40 + acol] = f2bf(ap[(size_t)it * 8 * K]);
    if (kb + 64 <= K) __builtin_prefetch(ap + 32, 0, 3);   // next K panel
  } else {                      // branchless: clamp address, select zero
    const int  kc  = imin(kb + acol, K - 1);
    const bool kok = (kb + acol < K);
#pragma unroll
    for (int it = 0; it < 16; ++it) {
      int r  = arow + it * 8;
      int rc = imin(r0 + r, M - 1);
      float v = A[(size_t)rc * K + kc];
      dst[r * 40 + acol] = f2bf((kok && (r0 + r < M)) ? v : 0.f);
    }
  }
}

// B tile: 128 n-rows x 32 k of pre-transposed bf16 weights -> async DMA to LDS.
// 512 16-byte segments, 2 per thread. ASYNCcnt-tracked; fenced by caller.
__device__ __forceinline__
void stageB_async(const unsigned short* __restrict__ Bt, unsigned short* dst,
                  int c0, int kb, int Kp, int tid) {
#pragma unroll
  for (int q = 0; q < 2; ++q) {
    int seg = tid * 2 + q;
    int n = seg >> 2, ks = (seg & 3) * 8;
    unsigned long long g =
        (unsigned long long)(size_t)(Bt + (size_t)(c0 + n) * Kp + kb + ks);
    unsigned l = (unsigned)(size_t)(dst + n * 40 + ks);
    asm volatile("global_load_async_to_lds_b128 %0, %1, off"
                 :: "v"(l), "v"(g) : "memory");
  }
}

// ---------------------------------------------------------------------------
// bf16-WMMA GEMM:  C[M,N] = act(A[M,K] @ W[K,N] + bias), W given as Bt[Npad,Kp]
// 256 threads = 8 wave32. Block tile 128x128, K-step 32, double-buffered LDS:
// stage tile k+1 (A: converting copy, B: async DMA) while 8 WMMAs/wave consume
// tile k; one barrier per K-step.
// ---------------------------------------------------------------------------
__global__ __launch_bounds__(256)
void gemm_wmma(const float* __restrict__ A, const unsigned short* __restrict__ Bt,
               const float* __restrict__ bias, float* __restrict__ C,
               int M, int N, int K, int Kp, int act) {
  __shared__ __align__(16) unsigned short As[2][128 * 40]; // [m][k] bf16
  __shared__ __align__(16) unsigned short Bs[2][128 * 40]; // [n][k] bf16

  const int tid  = threadIdx.x;
  const int lane = tid & 31;
  const int wv   = tid >> 5;
  const int r0 = blockIdx.y * 128;
  const int c0 = blockIdx.x * 128;
  const int mpair = wv & 3;       // two 16-row strips: rows mpair*32 .. +31
  const int nhalf = wv >> 2;      // 64 n-cols: nhalf*64 .. +63
  const bool fullM = (r0 + 128 <= M);
  const bool fullN = (c0 + 128 <= N);

  v8f acc[2][4];
#pragma unroll
  for (int t = 0; t < 2; ++t)
#pragma unroll
    for (int s = 0; s < 4; ++s)
#pragma unroll
      for (int j = 0; j < 8; ++j) acc[t][s][j] = 0.f;

  const int arow = tid >> 5;     // 0..7 (+8*it) ; lane = contiguous k
  const int acol = tid & 31;
  const int lm   = lane & 15;
  const int half = lane >> 4;

  // ---- prologue: stage tile 0 ----
  stageA(A, As[0], r0, 0, M, K, arow, acol, fullM);
  stageB_async(Bt, Bs[0], c0, 0, Kp, tid);
  asm volatile("s_wait_asynccnt 0" ::: "memory");
  __syncthreads();

  for (int kb = 0; kb < Kp; kb += 32) {
    const int cur = (kb >> 5) & 1;
    unsigned short* curA = As[cur];
    unsigned short* curB = Bs[cur];

    // ---- stage next tile into the other buffer ----
    if (kb + 32 < Kp) {
      stageA(A, As[cur ^ 1], r0, kb + 32, M, K, arow, acol, fullM);
      stageB_async(Bt, Bs[cur ^ 1], c0, kb + 32, Kp, tid);
    }

    // ---- fragments (two ds_load_b128 each) + 8 WMMAs ----
    v16bf af[2];
#pragma unroll
    for (int t = 0; t < 2; ++t) {
      int m_l = (mpair * 2 + t) * 16 + lm;
      *((uiv4*)&af[t])       = *(const uiv4*)&curA[m_l * 40 + half * 8];
      *(((uiv4*)&af[t]) + 1) = *(const uiv4*)&curA[m_l * 40 + 16 + half * 8];
    }
#pragma unroll
    for (int s = 0; s < 4; ++s) {
      int n_l = nhalf * 64 + s * 16 + lm;
      v16bf bf;
      *((uiv4*)&bf)       = *(const uiv4*)&curB[n_l * 40 + half * 8];
      *(((uiv4*)&bf) + 1) = *(const uiv4*)&curB[n_l * 40 + 16 + half * 8];
#pragma unroll
      for (int t = 0; t < 2; ++t)
        acc[t][s] = __builtin_amdgcn_wmma_f32_16x16x32_bf16(
            false, af[t], false, bf, (short)0, acc[t][s], false, false);
    }
    asm volatile("s_wait_asynccnt 0" ::: "memory");  // next B tile landed
    __syncthreads();
  }

  // ---- epilogue: C/D layout -> global, fused bias + ReLU ----
  if (fullM && fullN) {
#pragma unroll
    for (int s = 0; s < 4; ++s) {
      int n = c0 + nhalf * 64 + s * 16 + lm;
      float bv = bias ? bias[n] : 0.f;
#pragma unroll
      for (int t = 0; t < 2; ++t) {
        int mb = r0 + (mpair * 2 + t) * 16 + half * 8;
#pragma unroll
        for (int v = 0; v < 8; ++v) {
          float val = acc[t][s][v] + bv;
          if (act) val = fmaxf(val, 0.f);
          C[(size_t)(mb + v) * N + n] = val;
        }
      }
    }
  } else {
#pragma unroll
    for (int s = 0; s < 4; ++s) {
      int n = c0 + nhalf * 64 + s * 16 + lm;
      if (n >= N) continue;
      float bv = bias ? bias[n] : 0.f;
#pragma unroll
      for (int t = 0; t < 2; ++t) {
        int mb = r0 + (mpair * 2 + t) * 16 + half * 8;
#pragma unroll
        for (int v = 0; v < 8; ++v) {
          int m = mb + v;
          if (m < M) {
            float val = acc[t][s][v] + bv;
            if (act) val = fmaxf(val, 0.f);
            C[(size_t)m * N + n] = val;
          }
        }
      }
    }
  }
}

// ---------------------------------------------------------------------------
// WMMA 1-D convolution (kernel width 8), optionally fused with embedding:
//   tok != null: out[b,o,p] = relu(bias[o] + sum_{i<Ci,k<8} E[tok[b,i], p+k]*w[o,i,k])
//   tok == null: out[b,o,p] = relu(bias[o] + sum_{i<Ci,k<8} in[b,i,p+k]  *w[o,i,k])
// GEMM view: M=O (32-row tile), N=P (64-col tile), K=Ci*8 in 32-wide chunks.
// 8 waves = 2 m-strips x 4 n-tiles, 1 WMMA each. Branchless clamped staging.
// ---------------------------------------------------------------------------
__global__ __launch_bounds__(256)
void conv_wmma(const float* __restrict__ in,   // [B,Ci,Li] or embedding table [V,256]
               const int* __restrict__ tok,    // [B,Ci] tokens or nullptr
               const float* __restrict__ w,    // [O,Ci,8]
               const float* __restrict__ bias, // [O]
               float* __restrict__ out,        // [B,O,P]
               int Ci, int Li, int O, int P) {
  __shared__ __align__(16) unsigned short As[32 * 40];
  __shared__ __align__(16) unsigned short Bs[64 * 40];
  __shared__ int tks[768];

  const int b  = blockIdx.z;
  const int o0 = blockIdx.y * 32;
  const int p0 = blockIdx.x * 64;
  const int tid = threadIdx.x, lane = tid & 31, wv = tid >> 5;
  const int mw = wv >> 2;          // 0..1
  const int nw = wv & 3;           // 0..3

  if (tok)
    for (int i = tid; i < Ci; i += 256) tks[i] = tok[(size_t)b * Ci + i];
  __syncthreads();

  v8f acc;
#pragma unroll
  for (int j = 0; j < 8; ++j) acc[j] = 0.f;

  const int lm = lane & 15, half = lane >> 4;
  const int m_l = mw * 16 + lm;
  const int n_l = nw * 16 + lm;
  const int nchunks = (Ci + 3) >> 2;

  for (int ic = 0; ic < nchunks; ++ic) {
    const int ibase = ic * 4;
    // ---- stage A: weights, 32 o-rows x 32 (i,k) ----
    {
      int r = tid >> 3, kks = (tid & 7) * 4;
      int oc = imin(o0 + r, O - 1);
      bool ook = (o0 + r < O);
#pragma unroll
      for (int j = 0; j < 4; ++j) {
        int kk = kks + j;
        int i = ibase + (kk >> 3), k = kk & 7;
        int icl = imin(i, Ci - 1);
        float v = w[((size_t)oc * Ci + icl) * 8 + k];
        As[r * 40 + kk] = f2bf((ook && i < Ci) ? v : 0.f);
      }
    }
    // ---- stage B (transposed): 64 p-cols x 32 (i,k) ----
    {
      int n = tid >> 2, kks = (tid & 3) * 8;
      int p  = p0 + n;
      int pc = imin(p, P - 1);
      bool pok = (p < P);
#pragma unroll
      for (int j = 0; j < 8; ++j) {
        int kk = kks + j;
        int i = ibase + (kk >> 3), k = kk & 7;
        int icl = imin(i, Ci - 1);
        const float* row = tok ? (in + (size_t)tks[icl] * 256)
                               : (in + ((size_t)b * Ci + icl) * Li);
        float v = row[pc + k];          // pc+k <= P-1+7 <= row length-1
        Bs[n * 40 + kk] = f2bf((pok && i < Ci) ? v : 0.f);
      }
    }
    __syncthreads();

    v16bf afrag, bfrag;
    *((uiv4*)&afrag)       = *(const uiv4*)&As[m_l * 40 + half * 8];
    *(((uiv4*)&afrag) + 1) = *(const uiv4*)&As[m_l * 40 + 16 + half * 8];
    *((uiv4*)&bfrag)       = *(const uiv4*)&Bs[n_l * 40 + half * 8];
    *(((uiv4*)&bfrag) + 1) = *(const uiv4*)&Bs[n_l * 40 + 16 + half * 8];
    acc = __builtin_amdgcn_wmma_f32_16x16x32_bf16(
        false, afrag, false, bfrag, (short)0, acc, false, false);
    __syncthreads();
  }

  int p = p0 + nw * 16 + lm;
  if (p < P) {
#pragma unroll
    for (int v = 0; v < 8; ++v) {
      int o = o0 + mw * 16 + half * 8 + v;
      if (o < O)
        out[((size_t)b * O + o) * P + p] = fmaxf(acc[v] + bias[o], 0.f);
    }
  }
}

// ---------------------------------------------------------------------------
// Graph kernels
// ---------------------------------------------------------------------------
__global__ void build_edges(const int* __restrict__ ei, int* __restrict__ s2,
                            int* __restrict__ d2) {
  int i = blockIdx.x * blockDim.x + threadIdx.x;
  if (i >= E2) return;
  if (i < NE) { s2[i] = ei[i]; d2[i] = ei[NE + i]; }
  else        { s2[i] = i - NE; d2[i] = i - NE; }     // self loops
}

// es[n,h] = sum_c h[n,h*C+c]*asrc[h,c] ; ed likewise
__global__ void attn_scores(const float* __restrict__ h,
                            const float* __restrict__ asrc,
                            const float* __restrict__ adst,
                            float* __restrict__ es, float* __restrict__ ed,
                            int H, int C) {
  int idx = blockIdx.x * blockDim.x + threadIdx.x;
  if (idx >= NN * H) return;
  int n = idx / H, hh = idx % H;
  const float* hp = h + (size_t)n * H * C + (size_t)hh * C;
  const float* a1 = asrc + (size_t)hh * C;
  const float* a2 = adst + (size_t)hh * C;
  float s1 = 0.f, s2 = 0.f;
  for (int c = 0; c < C; ++c) { s1 += hp[c] * a1[c]; s2 += hp[c] * a2[c]; }
  es[idx] = s1; ed[idx] = s2;
}

__global__ void edge_max(const int* __restrict__ s2, const int* __restrict__ d2,
                         const float* __restrict__ es, const float* __restrict__ ed,
                         unsigned* __restrict__ mkey, int H) {
  int idx = blockIdx.x * blockDim.x + threadIdx.x;
  if (idx >= E2 * H) return;
  int e = idx / H, hh = idx % H;
  int s = s2[e], d = d2[e];
  float v = es[s * H + hh] + ed[d * H + hh];
  v = v > 0.f ? v : 0.2f * v;                       // LeakyReLU(0.2)
  atomicMax(&mkey[d * H + hh], fflip(v));
}

__global__ void edge_exp_sum(const int* __restrict__ s2, const int* __restrict__ d2,
                             const float* __restrict__ es, const float* __restrict__ ed,
                             const unsigned* __restrict__ mkey,
                             float* __restrict__ p, float* __restrict__ ssum, int H) {
  int idx = blockIdx.x * blockDim.x + threadIdx.x;
  if (idx >= E2 * H) return;
  int e = idx / H, hh = idx % H;
  int s = s2[e], d = d2[e];
  float v = es[s * H + hh] + ed[d * H + hh];
  v = v > 0.f ? v : 0.2f * v;
  float pe = expf(v - funflip(mkey[d * H + hh]));
  p[idx] = pe;
  atomicAdd(&ssum[d * H + hh], pe);
}

// out[dst, h*C+c] += alpha[e,h] * h[src, h*C+c]
__global__ void edge_aggregate(const int* __restrict__ s2, const int* __restrict__ d2,
                               const float* __restrict__ p, const float* __restrict__ ssum,
                               const float* __restrict__ h, float* __restrict__ out,
                               int H, int C) {
  __shared__ float al[16];
  int e = blockIdx.x;
  int s = s2[e], d = d2[e];
  if (threadIdx.x < H)
    al[threadIdx.x] = p[(size_t)e * H + threadIdx.x] / ssum[(size_t)d * H + threadIdx.x];
  __syncthreads();
  int F = H * C;
  const float* hs = h + (size_t)s * F;
  float* od = out + (size_t)d * F;
  for (int f = threadIdx.x; f < F; f += blockDim.x)
    atomicAdd(&od[f], al[f / C] * hs[f]);
}

__global__ void bias_relu(float* __restrict__ y, const float* __restrict__ b,
                          long long total, int F) {
  long long i = (long long)blockIdx.x * blockDim.x + threadIdx.x;
  if (i >= total) return;
  y[i] = fmaxf(y[i] + b[i % F], 0.f);
}

__global__ void deg_count(const int* __restrict__ d2, float* __restrict__ deg) {
  int i = blockIdx.x * blockDim.x + threadIdx.x;
  if (i < E2) atomicAdd(&deg[d2[i]], 1.f);
}

__global__ void gcn_aggregate(const int* __restrict__ s2, const int* __restrict__ d2,
                              const float* __restrict__ deg, const float* __restrict__ h,
                              float* __restrict__ out, int F) {
  __shared__ float wn;
  int e = blockIdx.x;
  int s = s2[e], d = d2[e];
  if (threadIdx.x == 0) wn = rsqrtf(deg[s]) * rsqrtf(deg[d]);
  __syncthreads();
  const float* hs = h + (size_t)s * F;
  float* od = out + (size_t)d * F;
  float w = wn;
  for (int f = threadIdx.x; f < F; f += blockDim.x)
    atomicAdd(&od[f], w * hs[f]);
}

// gmp/gap over NPG contiguous nodes per graph -> xg[g,0:F]=max, xg[g,F:2F]=mean
__global__ void graph_pool(const float* __restrict__ h, float* __restrict__ xg, int F) {
  int g = blockIdx.x;
  const float* base = h + (size_t)g * NPG * F;
  for (int f = threadIdx.x; f < F; f += blockDim.x) {
    float mx = -INFINITY, sm = 0.f;
    for (int i = 0; i < NPG; ++i) {
      float v = base[(size_t)i * F + f];
      mx = fmaxf(mx, v); sm += v;
    }
    xg[(size_t)g * 2 * F + f]     = mx;
    xg[(size_t)g * 2 * F + F + f] = sm * (1.f / NPG);
  }
}

// ---------------------------------------------------------------------------
// Pool / head kernels
// ---------------------------------------------------------------------------
__global__ void pool3(const float* __restrict__ in, float* __restrict__ out,
                      long long total, int Li, int Lo) {
  long long i = (long long)blockIdx.x * blockDim.x + threadIdx.x;
  if (i >= total) return;
  int q = (int)(i % Lo);
  long long bc = i / Lo;
  const float* p = in + bc * Li + (long long)q * 3;
  out[i] = fmaxf(fmaxf(p[0], p[1]), p[2]);
}

__global__ void concat3(const float* __restrict__ a, const float* __restrict__ b,
                        const float* __restrict__ c, float* __restrict__ xc) {
  int i = blockIdx.x * blockDim.x + threadIdx.x;
  if (i >= NB * 384) return;
  int bi = i / 384, ci = i % 384;
  float v;
  if (ci < 128)      v = a[bi * 128 + ci];
  else if (ci < 256) v = b[bi * 128 + ci - 128];
  else               v = c[bi * 128 + ci - 256];
  xc[i] = v;
}

__global__ void final_head(const float* __restrict__ h, const float* __restrict__ Wo,
                           const float* __restrict__ bo, float* __restrict__ out) {
  int b = blockIdx.x * blockDim.x + threadIdx.x;
  if (b >= NB) return;
  float acc = bo[0];
  const float* hp = h + (size_t)b * 128;
  for (int k = 0; k < 128; ++k) acc += hp[k] * Wo[k];
  out[b] = 1.f / (1.f + expf(-acc));
}

__global__ void copy_f32(const float* __restrict__ s, float* __restrict__ d, long long n) {
  long long i = (long long)blockIdx.x * blockDim.x + threadIdx.x;
  if (i < n) d[i] = s[i];
}

// ---------------------------------------------------------------------------
// Host
// ---------------------------------------------------------------------------
static inline unsigned g1(long long n, int t = 256) {
  return (unsigned)((n + t - 1) / t);
}

struct WtT { unsigned short* bt; int Kp; };

extern "C" void kernel_launch(void* const* d_in, const int* /*in_sizes*/, int /*n_in*/,
                              void* d_out, int /*out_size*/, void* d_ws, size_t /*ws_size*/,
                              hipStream_t stream) {
  const float* x    = (const float*)d_in[0];
  const int*   ei   = (const int*)d_in[1];
  const int*   tgt  = (const int*)d_in[3];
  const int*   drg  = (const int*)d_in[4];
  const float *W1=(const float*)d_in[5], *as1=(const float*)d_in[6], *ad1=(const float*)d_in[7], *b1=(const float*)d_in[8];
  const float *W2=(const float*)d_in[9], *as2=(const float*)d_in[10], *ad2=(const float*)d_in[11], *b2=(const float*)d_in[12];
  const float *W3=(const float*)d_in[13], *b3=(const float*)d_in[14];
  const float *Wg1=(const float*)d_in[15], *bg1=(const float*)d_in[16], *Wg2=(const float*)d_in[17], *bg2=(const float*)d_in[18];
  const float *Et=(const float*)d_in[19], *ct1w=(const float*)d_in[20], *ct1b=(const float*)d_in[21],
              *ct2w=(const float*)d_in[22], *ct2b=(const float*)d_in[23], *ct3w=(const float*)d_in[24], *ct3b=(const float*)d_in[25],
              *Wt1=(const float*)d_in[26], *bt1=(const float*)d_in[27], *Wt2=(const float*)d_in[28], *bt2=(const float*)d_in[29];
  const float *Ed=(const float*)d_in[30], *cd1w=(const float*)d_in[31], *cd1b=(const float*)d_in[32],
              *cd2w=(const float*)d_in[33], *cd2b=(const float*)d_in[34], *cd3w=(const float*)d_in[35], *cd3b=(const float*)d_in[36],
              *Wd1=(const float*)d_in[37], *bd1=(const float*)d_in[38], *Wd2=(const float*)d_in[39], *bd2=(const float*)d_in[40];
  const float *Wc1=(const float*)d_in[41], *bc1=(const float*)d_in[42], *Wc2=(const float*)d_in[43], *bc2=(const float*)d_in[44],
              *Wo=(const float*)d_in[45], *bo=(const float*)d_in[46];
  float* out = (float*)d_out;

  // ---- workspace (bump allocator, 256B aligned) ----
  size_t off = 0;
  auto alloc = [&](size_t nbytes) {
    void* p = (char*)d_ws + off;
    off += (nbytes + 255) & ~(size_t)255;
    return p;
  };
  int*   src2  = (int*)alloc((size_t)E2 * 4);
  int*   dst2  = (int*)alloc((size_t)E2 * 4);
  float* slabP = (float*)alloc((size_t)NN * 3120 * 4);  // h1 / h2pre / h3pre
  float* slabG = (float*)alloc((size_t)NN * 3120 * 4);  // g1/a1, g2/a2, g3/a3
  float* es    = (float*)alloc((size_t)NN * 10 * 4);
  float* edv   = (float*)alloc((size_t)NN * 10 * 4);
  unsigned* mkey = (unsigned*)alloc((size_t)NN * 10 * 4);
  float* ssum  = (float*)alloc((size_t)NN * 10 * 4);
  float* pbuf  = (float*)alloc((size_t)E2 * 10 * 4);
  float* deg   = (float*)alloc((size_t)NN * 4);
  float* xg    = (float*)alloc((size_t)NB * 6240 * 4);
  float* xg1   = (float*)alloc((size_t)NB * 1024 * 4);
  float* xg2   = (float*)alloc((size_t)NB * 128 * 4);
  float* t1    = (float*)alloc((size_t)NB * 32 * 249 * 4);
  float* tp1   = (float*)alloc((size_t)NB * 32 * 83 * 4);
  float* t2    = (float*)alloc((size_t)NB * 64 * 76 * 4);
  float* tp2   = (float*)alloc((size_t)NB * 64 * 25 * 4);
  float* t3    = (float*)alloc((size_t)NB * 128 * 18 * 4);
  float* tp3   = (float*)alloc((size_t)NB * 128 * 6 * 4);
  float* tf1   = (float*)alloc((size_t)NB * 1024 * 4);
  float* xt    = (float*)alloc((size_t)NB * 128 * 4);
  float* dd1   = (float*)alloc((size_t)NB * 25 * 249 * 4);
  float* dp1   = (float*)alloc((size_t)NB * 25 * 83 * 4);
  float* dd2   = (float*)alloc((size_t)NB * 50 * 76 * 4);
  float* dp2   = (float*)alloc((size_t)NB * 50 * 25 * 4);
  float* dd3   = (float*)alloc((size_t)NB * 100 * 18 * 4);
  float* dp3   = (float*)alloc((size_t)NB * 100 * 6 * 4);
  float* df1   = (float*)alloc((size_t)NB * 1024 * 4);
  float* xd    = (float*)alloc((size_t)NB * 128 * 4);
  float* xc    = (float*)alloc((size_t)NB * 384 * 4);
  float* c1    = (float*)alloc((size_t)NB * 1024 * 4);
  float* c2    = (float*)alloc((size_t)NB * 128 * 4);

  // ---- weight pre-pass: transpose + bf16 + pad, once per call ----
  auto mkwt = [&](const float* W, int K, int N) -> WtT {
    int Kp = (K + 31) & ~31;
    int Np = (N + 127) & ~127;
    unsigned short* bt = (unsigned short*)alloc((size_t)Np * Kp * 2);
    long long total = (long long)Np * Kp;
    wt_bf16t<<<g1(total), 256, 0, stream>>>(W, bt, K, N, Kp, total);
    return WtT{bt, Kp};
  };
  WtT tW1  = mkwt(W1,  78,   780);
  WtT tW2  = mkwt(W2,  780,  1560);
  WtT tW3  = mkwt(W3,  1560, 3120);
  WtT tWg1 = mkwt(Wg1, 6240, 1024);
  WtT tWg2 = mkwt(Wg2, 1024, 128);
  WtT tWt1 = mkwt(Wt1, 768,  1024);
  WtT tWt2 = mkwt(Wt2, 1024, 128);
  WtT tWd1 = mkwt(Wd1, 600,  1024);
  WtT tWd2 = mkwt(Wd2, 1024, 128);
  WtT tWc1 = mkwt(Wc1, 384,  1024);
  WtT tWc2 = mkwt(Wc2, 1024, 128);

  auto gemm = [&](const float* A, const WtT& Bt, const float* bias, float* C,
                  int M, int N, int K, int act) {
    dim3 grid((N + 127) / 128, (M + 127) / 128);
    gemm_wmma<<<grid, 256, 0, stream>>>(A, Bt.bt, bias, C, M, N, K, Bt.Kp, act);
  };
  auto conv = [&](const float* in, const int* tok, const float* w, const float* bias,
                  float* o, int Ci, int Li, int O, int P) {
    dim3 grid((P + 63) / 64, (O + 31) / 32, NB);
    conv_wmma<<<grid, 256, 0, stream>>>(in, tok, w, bias, o, Ci, Li, O, P);
  };

  // ---- edges with self loops ----
  build_edges<<<g1(E2), 256, 0, stream>>>(ei, src2, dst2);

  // ================= GAT layer 1 (H=10, C=78) =================
  gemm(x, tW1, nullptr, slabP, NN, 780, 78, 0);                      // h1
  attn_scores<<<g1((long long)NN * 10), 256, 0, stream>>>(slabP, as1, ad1, es, edv, 10, 78);
  fill_u32<<<g1((long long)NN * 10), 256, 0, stream>>>(mkey, 0u, (long long)NN * 10);
  fill_f32<<<g1((long long)NN * 10), 256, 0, stream>>>(ssum, 0.f, (long long)NN * 10);
  edge_max<<<g1((long long)E2 * 10), 256, 0, stream>>>(src2, dst2, es, edv, mkey, 10);
  edge_exp_sum<<<g1((long long)E2 * 10), 256, 0, stream>>>(src2, dst2, es, edv, mkey, pbuf, ssum, 10);
  fill_f32<<<g1((long long)NN * 780), 256, 0, stream>>>(slabG, 0.f, (long long)NN * 780);
  edge_aggregate<<<E2, 256, 0, stream>>>(src2, dst2, pbuf, ssum, slabP, slabG, 10, 78);
  bias_relu<<<g1((long long)NN * 780), 256, 0, stream>>>(slabG, b1, (long long)NN * 780, 780);

  // ================= GAT layer 2 (H=2, C=780) =================
  gemm(slabG, tW2, nullptr, slabP, NN, 1560, 780, 0);                // h2
  attn_scores<<<g1((long long)NN * 2), 256, 0, stream>>>(slabP, as2, ad2, es, edv, 2, 780);
  fill_u32<<<g1((long long)NN * 2), 256, 0, stream>>>(mkey, 0u, (long long)NN * 2);
  fill_f32<<<g1((long long)NN * 2), 256, 0, stream>>>(ssum, 0.f, (long long)NN * 2);
  edge_max<<<g1((long long)E2 * 2), 256, 0, stream>>>(src2, dst2, es, edv, mkey, 2);
  edge_exp_sum<<<g1((long long)E2 * 2), 256, 0, stream>>>(src2, dst2, es, edv, mkey, pbuf, ssum, 2);
  fill_f32<<<g1((long long)NN * 1560), 256, 0, stream>>>(slabG, 0.f, (long long)NN * 1560);
  edge_aggregate<<<E2, 256, 0, stream>>>(src2, dst2, pbuf, ssum, slabP, slabG, 2, 780);
  bias_relu<<<g1((long long)NN * 1560), 256, 0, stream>>>(slabG, b2, (long long)NN * 1560, 1560);

  // ================= GCN layer =================
  gemm(slabG, tW3, nullptr, slabP, NN, 3120, 1560, 0);               // h3
  fill_f32<<<g1(NN), 256, 0, stream>>>(deg, 0.f, NN);
  deg_count<<<g1(E2), 256, 0, stream>>>(dst2, deg);
  fill_f32<<<g1((long long)NN * 3120), 256, 0, stream>>>(slabG, 0.f, (long long)NN * 3120);
  gcn_aggregate<<<E2, 256, 0, stream>>>(src2, dst2, deg, slabP, slabG, 3120);
  bias_relu<<<g1((long long)NN * 3120), 256, 0, stream>>>(slabG, b3, (long long)NN * 3120, 3120);

  // ================= graph pooling + graph FC =================
  graph_pool<<<NB, 256, 0, stream>>>(slabG, xg, 3120);
  gemm(xg,  tWg1, bg1, xg1, NB, 1024, 6240, 1);
  gemm(xg1, tWg2, bg2, xg2, NB, 128, 1024, 1);
  copy_f32<<<g1((long long)NB * 128), 256, 0, stream>>>(xg2, out + NB, (long long)NB * 128);

  // ================= target CNN branch (WMMA convs) =================
  conv(Et, tgt, ct1w, ct1b, t1, 735, 256, 32, 249);
  pool3<<<g1((long long)NB * 32 * 83), 256, 0, stream>>>(t1, tp1, (long long)NB * 32 * 83, 249, 83);
  conv(tp1, nullptr, ct2w, ct2b, t2, 32, 83, 64, 76);
  pool3<<<g1((long long)NB * 64 * 25), 256, 0, stream>>>(t2, tp2, (long long)NB * 64 * 25, 76, 25);
  conv(tp2, nullptr, ct3w, ct3b, t3, 64, 25, 128, 18);
  pool3<<<g1((long long)NB * 128 * 6), 256, 0, stream>>>(t3, tp3, (long long)NB * 128 * 6, 18, 6);
  gemm(tp3, tWt1, bt1, tf1, NB, 1024, 768, 1);
  gemm(tf1, tWt2, bt2, xt, NB, 128, 1024, 0);

  // ================= drug CNN branch (WMMA convs) =================
  conv(Ed, drg, cd1w, cd1b, dd1, 100, 256, 25, 249);
  pool3<<<g1((long long)NB * 25 * 83), 256, 0, stream>>>(dd1, dp1, (long long)NB * 25 * 83, 249, 83);
  conv(dp1, nullptr, cd2w, cd2b, dd2, 25, 83, 50, 76);
  pool3<<<g1((long long)NB * 50 * 25), 256, 0, stream>>>(dd2, dp2, (long long)NB * 50 * 25, 76, 25);
  conv(dp2, nullptr, cd3w, cd3b, dd3, 50, 25, 100, 18);
  pool3<<<g1((long long)NB * 100 * 6), 256, 0, stream>>>(dd3, dp3, (long long)NB * 100 * 6, 18, 6);
  gemm(dp3, tWd1, bd1, df1, NB, 1024, 600, 1);
  gemm(df1, tWd2, bd2, xd, NB, 128, 1024, 0);

  // ================= head =================
  concat3<<<g1((long long)NB * 384), 256, 0, stream>>>(xg2, xd, xt, xc);
  gemm(xc, tWc1, bc1, c1, NB, 1024, 384, 1);
  gemm(c1, tWc2, bc2, c2, NB, 128, 1024, 1);
  final_head<<<g1(NB), 256, 0, stream>>>(c2, Wo, bo, out);
}
